// VectorQuantizer_14989435863664
// MI455X (gfx1250) — matrix operand order repaired and checked by reading, not compiled, not run on hardware
//
#include <hip/hip_runtime.h>

typedef __attribute__((ext_vector_type(16))) _Float16 v16h;
typedef __attribute__((ext_vector_type(8)))  float    v8f;

#define B_      8
#define D_      64
#define T_      8192
#define K_      1024
#define TILE_T  128
#define NTILE   (T_ / TILE_T)          // 64 tiles per batch
#define NBLK    (B_ * NTILE)           // 512 workgroups
#define NELEM   (B_ * D_ * T_)         // 4194304

// d_ws layout (bytes)
#define WS_CBB_OFF   0                         // 65536 halves  = 131072 B (B-fragment f16 codebook)
#define WS_E2_OFF    131072                    // 1024 floats   =   4096 B (||e_k||^2 + 1)
#define WS_LOSS_OFF  (131072 + 4096)           // 512 floats (per-block loss partials)

// ---------------------------------------------------------------------------
// Kernel 0: precompute ||e_k||^2 + 1 (bias folded for the u32 argmin-key
// trick) and swizzle codebook (f32 -> f16) into a WMMA B-fragment
// layout for V_WMMA_F32_16X16X32_F16 in this file's contiguous k order (same as the A staging).
// B chunk = 32(K=d) x 16(N=code). Lane = n + (dd<16?0:16), half j = dd&15.
// Fragment (colChunk c, kChunk h) at halves [((c*2+h)*32+lane)*16 .. +15]:
// each lane loads one contiguous 32-byte v16h.
// ---------------------------------------------------------------------------
__global__ void vq_prep_codebook(const float* __restrict__ codebook,
                                 _Float16* __restrict__ cbB,
                                 float* __restrict__ e2p) {
    int k = blockIdx.x * blockDim.x + threadIdx.x;
    if (k >= K_) return;
    const float* row = codebook + k * D_;
    float sumsq = 0.0f;
    int c = k >> 4;          // column chunk (16 codes)
    int n = k & 15;          // code within chunk
#pragma unroll
    for (int d = 0; d < D_; ++d) {
        float v = row[d];
        sumsq += v * v;
        int h    = d >> 5;                 // k-dim chunk of 32
        int dd   = d & 31;
        int lane = n + ((dd >> 4) << 4);   // lanes 0-15: K 0-15, lanes 16-31: K 16-31
        int j    = dd & 15;
        cbB[(((c * 2 + h) * 32 + lane) << 4) + j] = (_Float16)v;
    }
    e2p[k] = sumsq + 1.0f;   // +1 bias keeps dist keys positive (|2 z.e| < 0.25 here)
}

// ---------------------------------------------------------------------------
// Kernel 1: main VQ kernel. One workgroup = 256 threads = 8 waves handles a
// tile of 128 t-values for one batch b. Each wave owns 16 rows (A fragments
// in registers) and sweeps all 1024 codes, two 16-column chunks per
// iteration (2 accumulators x 2 WMMA) so the WMMA->VALU hazard windows of
// one chain are filled by the other chain's work.
// Argmin: biased distance (positive float => unsigned-comparable bits) with
// the 6-bit chunk id packed into the low mantissa bits (v_bfi + v_min_u32).
// ---------------------------------------------------------------------------
__global__ __launch_bounds__(256)
void vq_main(const float* __restrict__ z,        // [B, D, T]
             const float* __restrict__ codebook, // [K, D] f32 (for exact gather)
             const _Float16* __restrict__ cbB,   // swizzled f16 B fragments
             const float* __restrict__ e2g,      // ||e_k||^2 + 1
             float* __restrict__ out_codes,      // [B*T] (as float)
             float* __restrict__ out_q,          // [B, D, T]
             float* __restrict__ loss_part) {    // [NBLK]
    __shared__ __align__(32) _Float16 smemA[8 * 2 * 32 * 16]; // 16 KB A fragments
    __shared__ float        smemE2[K_];                       // 4 KB
    __shared__ unsigned int smemMinK[8 * 32 * 8];             // 8 KB partial argmin keys
    __shared__ int          smemCode[TILE_T];                 // 512 B
    __shared__ float        smemLoss[256];                    // 1 KB

    const int tid  = threadIdx.x;
    const int blk  = blockIdx.x;
    const int b    = blk >> 6;            // blk / NTILE
    const int t0   = (blk & 63) * TILE_T;
    const int lane = tid & 31;
    const int w    = tid >> 5;            // wave id 0..7 (wave32)

    // ---- stage z tile [64 d x 128 t] -> LDS in A-fragment f16 layout ----
    // 16-bit A 16x32: lanes 0-15 hold K(d%32) in {0..15}, lanes 16-31 in {16..31} (matches cbB)
    const size_t zbase = (size_t)b * (D_ * T_) + t0;
#pragma unroll
    for (int p = 0; p < 8; ++p) {
        int linear = p * 1024 + tid * 4;         // over 64*128 elements
        int d  = linear >> 7;
        int tl = linear & 127;
        float4 v = *(const float4*)(z + zbase + (size_t)d * T_ + tl);
        int h  = d >> 5;
        int dd = d & 31;
        int lA = (dd >> 4) << 4;                       // +16 lanes for dd >= 16
        int jA = dd & 15;                              // half index = dd within 16
        float vv[4] = {v.x, v.y, v.z, v.w};
#pragma unroll
        for (int i = 0; i < 4; ++i) {
            int tt = tl + i;
            int r  = tt >> 4;
            int m  = tt & 15;
            smemA[(((r * 2 + h) * 32 + (m + lA)) << 4) + jA] = (_Float16)vv[i];
        }
    }
    // ---- stage ||e||^2 + 1 -> LDS ----
#pragma unroll
    for (int p = 0; p < 4; ++p) smemE2[p * 256 + tid] = e2g[p * 256 + tid];
    __syncthreads();

    // ---- A fragments for this wave (16 rows), K-chunks h=0,1 ----
    v16h a0 = *(const v16h*)&smemA[((w * 2 + 0) * 32 + lane) << 4];
    v16h a1 = *(const v16h*)&smemA[((w * 2 + 1) * 32 + lane) << 4];

    const int nLane = lane & 15;              // column-within-chunk (constant per lane)
    unsigned int mink[8];
#pragma unroll
    for (int q = 0; q < 8; ++q) mink[q] = 0xFFFFFFFFu;

    // ---- sweep all 1024 codes, two 16-column chunks per iteration ----
    for (int c2 = 0; c2 < (K_ / 32); ++c2) {
        const int c0 = 2 * c2, c1 = 2 * c2 + 1;
        const _Float16* bb = cbB + (((c0 * 2) * 32 + lane) << 4);
        v16h b0 = *(const v16h*)(bb);             // chunk c0, k 0-31
        v16h b1 = *(const v16h*)(bb + 512);       // chunk c0, k 32-63
        v16h b2 = *(const v16h*)(bb + 1024);      // chunk c1, k 0-31
        v16h b3 = *(const v16h*)(bb + 1536);      // chunk c1, k 32-63
        if (c2 + 1 < (K_ / 32))
            __builtin_prefetch(bb + 2048, 0, 0);

        v8f accA = {}, accB = {};
        accA = __builtin_amdgcn_wmma_f32_16x16x32_f16(false, a0, false, b0,
                                                      (short)0, accA, false, false);
        accA = __builtin_amdgcn_wmma_f32_16x16x32_f16(false, a1, false, b1,
                                                      (short)0, accA, false, false);
        accB = __builtin_amdgcn_wmma_f32_16x16x32_f16(false, a0, false, b2,
                                                      (short)0, accB, false, false);
        accB = __builtin_amdgcn_wmma_f32_16x16x32_f16(false, a1, false, b3,
                                                      (short)0, accB, false, false);

        float e2A = smemE2[(c0 << 4) + nLane];
        float e2B = smemE2[(c1 << 4) + nLane];
#pragma unroll
        for (int q = 0; q < 8; ++q) {             // key = bits(1+dist) with chunk id in low 6
            float dp = __builtin_fmaf(-2.0f, accA[q], e2A);
            unsigned int key = (__float_as_uint(dp) & ~63u) | ((unsigned)c0 & 63u);
            mink[q] = key < mink[q] ? key : mink[q];
        }
#pragma unroll
        for (int q = 0; q < 8; ++q) {
            float dp = __builtin_fmaf(-2.0f, accB[q], e2B);
            unsigned int key = (__float_as_uint(dp) & ~63u) | ((unsigned)c1 & 63u);
            mink[q] = key < mink[q] ? key : mink[q];
        }
    }

    // ---- cross-lane argmin reduction via LDS ----
#pragma unroll
    for (int q = 0; q < 8; ++q)
        smemMinK[(w * 32 + lane) * 8 + q] = mink[q];
    __syncthreads();

    if (tid < TILE_T) {
        int rowLocal = tid;
        int w2   = rowLocal >> 4;
        int m    = rowLocal & 15;
        int slot = m & 7;
        int lb   = (m >> 3) << 4;      // rows 0-7 in lanes 0-15, rows 8-15 in lanes 16-31
        unsigned int bestd = 0xFFFFFFFFu; int bestcol = 0;
#pragma unroll
        for (int l = 0; l < 16; ++l) {
            unsigned int k  = smemMinK[(w2 * 32 + (lb + l)) * 8 + slot];
            unsigned int md = k & ~63u;                 // quantized distance bits
            int col = (int)((k & 63u) << 4) | l;        // chunk*16 + lane
            if (md < bestd || (md == bestd && col < bestcol)) { bestd = md; bestcol = col; }
        }
        smemCode[rowLocal] = bestcol;
        out_codes[(size_t)b * T_ + t0 + rowLocal] = (float)bestcol;
    }
    __syncthreads();

    // ---- gather quantized (exact f32 codebook), write out, accumulate loss ----
    float lossp = 0.0f;
#pragma unroll
    for (int p = 0; p < 32; ++p) {
        int idx = p * 256 + tid;       // over 64*128 tile elements
        int d   = idx >> 7;
        int tl  = idx & 127;
        int code = smemCode[tl];
        float qv = codebook[code * D_ + d];
        float zv = z[zbase + (size_t)d * T_ + tl];
        out_q[zbase + (size_t)d * T_ + tl] = qv;   // straight-through fwd == quantized
        float diff = qv - zv;
        lossp += diff * diff;
    }
    smemLoss[tid] = lossp;
    __syncthreads();
#pragma unroll
    for (int s = 128; s > 0; s >>= 1) {
        if (tid < s) smemLoss[tid] += smemLoss[tid + s];
        __syncthreads();
    }
    if (tid == 0) loss_part[blk] = smemLoss[0];
}

// ---------------------------------------------------------------------------
// Kernel 2: reduce per-block loss partials -> scalar commit loss (mean)
// ---------------------------------------------------------------------------
__global__ void vq_loss_reduce(const float* __restrict__ loss_part,
                               float* __restrict__ out_loss) {
    __shared__ float s[256];
    int tid = threadIdx.x;
    s[tid] = loss_part[tid] + loss_part[tid + 256];
    __syncthreads();
#pragma unroll
    for (int st = 128; st > 0; st >>= 1) {
        if (tid < st) s[tid] += s[tid + st];
        __syncthreads();
    }
    if (tid == 0) out_loss[0] = s[0] * (1.0f / (float)NELEM);
}

extern "C" void kernel_launch(void* const* d_in, const int* in_sizes, int n_in,
                              void* d_out, int out_size, void* d_ws, size_t ws_size,
                              hipStream_t stream) {
    const float* z        = (const float*)d_in[0];   // [B, D, T]
    const float* codebook = (const float*)d_in[1];   // [K, D]

    char* ws = (char*)d_ws;
    _Float16* cbB   = (_Float16*)(ws + WS_CBB_OFF);
    float*    e2p   = (float*)(ws + WS_E2_OFF);
    float*    lossp = (float*)(ws + WS_LOSS_OFF);

    float* outF      = (float*)d_out;
    float* out_codes = outF;                       // B*T
    float* out_q     = outF + (size_t)B_ * T_;     // B*D*T
    float* out_loss  = out_q + (size_t)NELEM;      // 1

    vq_prep_codebook<<<K_ / 256, 256, 0, stream>>>(codebook, cbB, e2p);
    vq_main<<<NBLK, 256, 0, stream>>>(z, codebook, cbB, e2p,
                                      out_codes, out_q, lossp);
    vq_loss_reduce<<<1, 256, 0, stream>>>(lossp, out_loss);
}
